// SubgraphMining_26061861552901
// MI455X (gfx1250) — matrix operand order repaired and checked by reading, not compiled, not run on hardware
//
#include <hip/hip_runtime.h>
#include <hip/hip_bf16.h>

// ---------------- WMMA types ----------------
typedef __attribute__((ext_vector_type(16))) __bf16 v16bf;
typedef __attribute__((ext_vector_type(8)))  __bf16 v8bf;
typedef __attribute__((ext_vector_type(8)))  float  v8f;

#define H_DIM   256
#define K_TOT   768
#define M_DIM   64
#define NGRAPH  128

// LDS strides (bf16 elements), padded for conflict-free half-wave access:
//  stride 40 -> 20 dwords -> banks (20*c)%64 distinct for c=0..15, 16B aligned
//  stride 264 -> 132 dwords -> banks (4*c)%64 distinct, 16B aligned
#define A_STR   40
#define W_STR   40
#define H_STR   264
#define W2_STR  40

#define SH_A    0                       // 64  x A_STR  = 2560
#define SH_W    2560                    // 256 x W_STR  = 10240  (phase 1 total 12800)
#define SH_H    0                       // 64  x H_STR  = 16896
#define SH_W2   16896                   // 64  x W2_STR = 2560
#define SH_TOT  19456                   // 38912 bytes

// A-fragment (16x32 bf16) from an LDS row of 32 contiguous K values.
// ISA layout: lanes 0-15 (M=lane): K0..7 in v0-3, K16..23 in v4-7.
//             lanes 16-31 (M=lane-16): K8..15, K24..31.
__device__ __forceinline__ v16bf load_a16(const __bf16* rowPtr, int half) {
  v8bf lo = *(const v8bf*)(rowPtr + half * 8);
  v8bf hi = *(const v8bf*)(rowPtr + 16 + half * 8);
  return __builtin_shufflevector(lo, hi, 0,1,2,3,4,5,6,7,8,9,10,11,12,13,14,15);
}

// B-fragment (32x16 bf16), LDS staged col-major: colPtr = 32 contiguous K.
// lanes 0-15: col=lane, K0..7, K16..23 ; lanes 16-31: col=lane-16, K8..15, K24..31.
__device__ __forceinline__ v16bf load_b16(const __bf16* colPtr, int half) {
  v8bf lo = *(const v8bf*)(colPtr + half * 8);
  v8bf hi = *(const v8bf*)(colPtr + 16 + half * 8);
  return __builtin_shufflevector(lo, hi, 0,1,2,3,4,5,6,7,8,9,10,11,12,13,14,15);
}

// ---- CDNA5 async global->LDS copy (no VGPR round-trip; ASYNCcnt tracked) ----
// Generic LDS pointer: low 32 bits == LDS byte offset (ISA flat-aperture rule).
__device__ __forceinline__ unsigned lds_off(const void* p) {
  return (unsigned)(unsigned long long)p;
}
// 64-byte copy: IOFFSET applies to both LDS and global address (ISA 08 §4.4).
__device__ __forceinline__ void async_copy64(unsigned lds, const void* g) {
  asm volatile("global_load_async_to_lds_b128 %0, %1, off\n\t"
               "global_load_async_to_lds_b128 %0, %1, off offset:16\n\t"
               "global_load_async_to_lds_b128 %0, %1, off offset:32\n\t"
               "global_load_async_to_lds_b128 %0, %1, off offset:48"
               :: "v"(lds), "v"(g) : "memory");
}
__device__ __forceinline__ void async_copy16(unsigned lds, const void* g) {
  asm volatile("global_load_async_to_lds_b128 %0, %1, off"
               :: "v"(lds), "v"(g) : "memory");
}
__device__ __forceinline__ void wait_async0() {
  asm volatile("s_wait_asynccnt 0x0" ::: "memory");
}

// ---------------- prep kernels ----------------
__global__ void gnn_w1_prep(const float* __restrict__ W1, __bf16* __restrict__ W1t) {
  int idx = blockIdx.x * 256 + threadIdx.x;          // 768*256 elements
  if (idx < K_TOT * H_DIM) {
    int n = idx / K_TOT, k = idx % K_TOT;
    W1t[(size_t)n * K_TOT + k] = (__bf16)W1[(size_t)k * H_DIM + n];
  }
}

__global__ void gnn_w2_prep(const float* __restrict__ W2, __bf16* __restrict__ W2t) {
  int idx = blockIdx.x * 256 + threadIdx.x;          // 256*64 elements
  if (idx < H_DIM * M_DIM) {
    int n = idx / H_DIM, k = idx % H_DIM;
    W2t[(size_t)n * H_DIM + k] = (__bf16)W2[(size_t)k * M_DIM + n];
  }
}

__global__ void gnn_degree(const int* __restrict__ ei, float* __restrict__ deg, int E) {
  int e = blockIdx.x * 256 + threadIdx.x;
  if (e < E) atomicAdd(deg + ei[e], 1.0f);           // src row of edge_index
}

__global__ void gnn_counts(const int* __restrict__ batch, float* __restrict__ gcnt, int N) {
  int i = blockIdx.x * 256 + threadIdx.x;
  if (i < N) atomicAdd(gcnt + batch[i], 1.0f);
}

// ---------------- edge scatter: nsum[src] += x[dst] ----------------
__global__ void gnn_scatter(const float* __restrict__ x, const int* __restrict__ ei,
                            float* __restrict__ nsum, int E) {
  long long idx = (long long)blockIdx.x * 256 + threadIdx.x;   // E*64 threads
  int e  = (int)(idx >> 6);
  int c4 = ((int)idx & 63) * 4;
  if (e < E) {
    int s = ei[e];
    int d = ei[E + e];
    const float* xr = x    + (size_t)d * H_DIM + c4;
    float*       o  = nsum + (size_t)s * H_DIM + c4;
    atomicAdd(o + 0, xr[0]);
    atomicAdd(o + 1, xr[1]);
    atomicAdd(o + 2, xr[2]);
    atomicAdd(o + 3, xr[3]);
  }
}

// ---------------- agg = nsum / deg (in place) ----------------
__global__ void gnn_aggdiv(float* __restrict__ nsum, const float* __restrict__ deg, long long total) {
  long long idx = (long long)blockIdx.x * 256 + threadIdx.x;
  if (idx < total) {
    float d = deg[idx >> 8];                         // /H_DIM
    float v = nsum[idx];
    nsum[idx] = (d > 0.f) ? (v / d) : 0.f;
  }
}

// ---------------- fused MLP + pooling (WMMA bf16) ----------------
// One K-chunk of GEMM1 for a fixed feature segment (0:x 1:agg 2:x*agg).
template <int SEG>
__device__ __forceinline__ void gemm1_chunk(
    __bf16* shA, __bf16* shW, const __bf16* W1t,
    const float* xrow, const float* grow,
    int k0, int t, int arow, int acol, int rt, int cb, int l16, int half,
    v8f acc[8])
{
  const int kb = (k0 & 255) + acol;
  __syncthreads();
  // stage W1 chunk via async DMA: thread t == column, 64B contiguous
  async_copy64(lds_off(shW + t * W_STR), W1t + (size_t)t * K_TOT + k0);
  {   // stage feat chunk (bf16) -- needs VALU conversion, so classic ds path
    v8bf av;
    #pragma unroll
    for (int i = 0; i < 8; ++i) {
      float v;
      if      (SEG == 0) v = xrow[kb + i];
      else if (SEG == 1) v = grow[kb + i];
      else               v = xrow[kb + i] * grow[kb + i];
      av[i] = (__bf16)v;
    }
    *(v8bf*)(shA + arow * A_STR + acol) = av;
  }
  wait_async0();
  __syncthreads();

  v16bf af = load_a16(shA + (rt * 16 + l16) * A_STR, half);
  #pragma unroll
  for (int g = 0; g < 2; ++g) {         // 2 groups of 4: preload frags, then 4 back-to-back WMMAs
    v16bf bfr[4];
    #pragma unroll
    for (int c = 0; c < 4; ++c)
      bfr[c] = load_b16(shW + ((cb + g * 4 + c) * 16 + l16) * W_STR, half);
    #pragma unroll
    for (int c = 0; c < 4; ++c)
      acc[g * 4 + c] = __builtin_amdgcn_wmma_f32_16x16x32_bf16(
          false, af, false, bfr[c], (short)0, acc[g * 4 + c], false, false);
  }
}

// 64 nodes per workgroup, 256 threads (8 waves).
__global__ __launch_bounds__(256) void gnn_fused_mlp(
    const float* __restrict__ x, const float* __restrict__ agg,
    const __bf16* __restrict__ W1t, const __bf16* __restrict__ W2t,
    const float* __restrict__ b1, const float* __restrict__ b2,
    const int* __restrict__ batch, float* __restrict__ gsum, int N)
{
  __shared__ __bf16 sh[SH_TOT];
  __bf16* shA  = sh + SH_A;               // phase1: 64 x A_STR feat
  __bf16* shW  = sh + SH_W;               // phase1: 256 cols x W_STR (col-major)
  __bf16* shH  = sh + SH_H;               // phase2: 64 x H_STR h
  __bf16* shW2 = sh + SH_W2;              // phase2: 64 cols x W2_STR

  const int t    = threadIdx.x;
  const int wid  = t >> 5;
  const int lane = t & 31;
  const int half = lane >> 4;
  const int l16  = lane & 15;
  const int rt   = wid >> 1;              // row tile 0..3
  const int cb   = (wid & 1) * 8;         // GEMM1 col-tile base (of 16)
  const int tileBase = blockIdx.x * 64;

  // staging coords for feat: each thread owns 8 contiguous cols of one row
  const int arow = t >> 2;                // 0..63
  const int acol = (t & 3) * 8;
  int rowg = tileBase + arow; if (rowg >= N) rowg = N - 1;   // clamp, masked later
  const float* xrow = x   + (size_t)rowg * H_DIM;
  const float* grow = agg + (size_t)rowg * H_DIM;

  // ---- GEMM1: feat[64x768] @ W1[768x256], K chunks of 32, 3 segments ----
  v8f acc[8] = {};
  for (int kc = 0; kc < 8; ++kc)
    gemm1_chunk<0>(shA, shW, W1t, xrow, grow, kc * 32,       t, arow, acol, rt, cb, l16, half, acc);
  for (int kc = 0; kc < 8; ++kc)
    gemm1_chunk<1>(shA, shW, W1t, xrow, grow, 256 + kc * 32, t, arow, acol, rt, cb, l16, half, acc);
  for (int kc = 0; kc < 8; ++kc)
    gemm1_chunk<2>(shA, shW, W1t, xrow, grow, 512 + kc * 32, t, arow, acol, rt, cb, l16, half, acc);

  // ---- bias + relu + bf16 h into LDS ----
  __syncthreads();
  #pragma unroll
  for (int c = 0; c < 8; ++c) {
    const int colg = (cb + c) * 16 + l16;
    const float bias = b1[colg];
    #pragma unroll
    for (int r = 0; r < 8; ++r) {
      float v = acc[c][r] + bias;
      v = v > 0.f ? v : 0.f;
      const int row = rt * 16 + half * 8 + r;
      shH[row * H_STR + colg] = (__bf16)v;
    }
  }
  __syncthreads();

  // ---- GEMM2: h[64x256] @ W2[256x64] ----
  v8f acc2[2] = {};
  const int ct0 = (wid & 1) * 2;          // col tiles 0..3, two per wave
  for (int kc = 0; kc < H_DIM / 32; ++kc) {
    const int k0 = kc * 32;
    __syncthreads();
    {   // stage W2 chunk (64 cols x 32 k, col-major) via async DMA: 16B/thread
      const int col = t >> 2;
      const int kk  = (t & 3) * 8;
      async_copy16(lds_off(shW2 + col * W2_STR + kk),
                   W2t + (size_t)col * H_DIM + k0 + kk);
    }
    wait_async0();
    __syncthreads();

    v16bf af = load_a16(shH + (rt * 16 + l16) * H_STR + k0, half);
    v16bf bf0 = load_b16(shW2 + ((ct0 + 0) * 16 + l16) * W2_STR, half);
    v16bf bf1 = load_b16(shW2 + ((ct0 + 1) * 16 + l16) * W2_STR, half);
    acc2[0] = __builtin_amdgcn_wmma_f32_16x16x32_bf16(
                false, af, false, bf0, (short)0, acc2[0], false, false);
    acc2[1] = __builtin_amdgcn_wmma_f32_16x16x32_bf16(
                false, af, false, bf1, (short)0, acc2[1], false, false);
  }

  // ---- b2 + per-graph pooling (atomics; divergence only after all WMMA) ----
  #pragma unroll
  for (int c = 0; c < 2; ++c) {
    const int colg = (ct0 + c) * 16 + l16;
    const float bias = b2[colg];
    #pragma unroll
    for (int r = 0; r < 8; ++r) {
      const int row = tileBase + rt * 16 + half * 8 + r;
      if (row < N) {
        const int g = batch[row];
        atomicAdd(gsum + g * M_DIM + colg, acc2[c][r] + bias);
      }
    }
  }
}

// ---------------- final divide ----------------
__global__ void gnn_final(const float* __restrict__ gsum, const float* __restrict__ gcnt,
                          float* __restrict__ out) {
  int idx = blockIdx.x * 256 + threadIdx.x;          // 128*64
  if (idx < NGRAPH * M_DIM) {
    float c = gcnt[idx >> 6];
    out[idx] = gsum[idx] / (c > 1.f ? c : 1.f);
  }
}

// ---------------- launch ----------------
static inline size_t alignUp(size_t v, size_t a) { return (v + a - 1) & ~(a - 1); }

extern "C" void kernel_launch(void* const* d_in, const int* in_sizes, int n_in,
                              void* d_out, int out_size, void* d_ws, size_t ws_size,
                              hipStream_t stream) {
  const float* x     = (const float*)d_in[0];
  const int*   ei    = (const int*)  d_in[1];
  const int*   batch = (const int*)  d_in[2];
  const float* W1    = (const float*)d_in[3];
  const float* b1    = (const float*)d_in[4];
  const float* W2    = (const float*)d_in[5];
  const float* b2    = (const float*)d_in[6];
  float* out = (float*)d_out;

  const int N = in_sizes[0] / H_DIM;
  const int E = in_sizes[1] / 2;

  char* ws = (char*)d_ws;
  size_t offNsum = 0;
  size_t offDeg  = offNsum + (size_t)N * H_DIM * 4;
  size_t offGsum = alignUp(offDeg + (size_t)N * 4, 256);
  size_t offGcnt = offGsum + (size_t)NGRAPH * M_DIM * 4;
  size_t zeroEnd = offGcnt + (size_t)NGRAPH * 4;
  size_t offW1t  = alignUp(zeroEnd, 256);
  size_t offW2t  = alignUp(offW1t + (size_t)K_TOT * H_DIM * 2, 256);

  float*  nsum = (float*) (ws + offNsum);
  float*  deg  = (float*) (ws + offDeg);
  float*  gsum = (float*) (ws + offGsum);
  float*  gcnt = (float*) (ws + offGcnt);
  __bf16* W1t  = (__bf16*)(ws + offW1t);
  __bf16* W2t  = (__bf16*)(ws + offW2t);

  // zero accumulators (graph-capture-safe)
  hipMemsetAsync(ws, 0, zeroEnd, stream);

  // weight transpose + bf16
  gnn_w1_prep<<<(K_TOT * H_DIM + 255) / 256, 256, 0, stream>>>(W1, W1t);
  gnn_w2_prep<<<(H_DIM * M_DIM + 255) / 256, 256, 0, stream>>>(W2, W2t);

  // degrees + per-graph counts
  gnn_degree<<<(E + 255) / 256, 256, 0, stream>>>(ei, deg, E);
  gnn_counts<<<(N + 255) / 256, 256, 0, stream>>>(batch, gcnt, N);

  // edge scatter
  long long sThreads = (long long)E * (H_DIM / 4);
  gnn_scatter<<<(unsigned)((sThreads + 255) / 256), 256, 0, stream>>>(x, ei, nsum, E);

  // agg = nsum / deg
  long long aTotal = (long long)N * H_DIM;
  gnn_aggdiv<<<(unsigned)((aTotal + 255) / 256), 256, 0, stream>>>(nsum, deg, aTotal);

  // fused MLP + pooling
  gnn_fused_mlp<<<(N + 63) / 64, 256, 0, stream>>>(x, nsum, W1t, W2t, b1, b2, batch, gsum, N);

  // final mean
  gnn_final<<<(NGRAPH * M_DIM + 255) / 256, 256, 0, stream>>>(gsum, gcnt, out);
}
